// MoEFusedTKG_16088947491299
// MI455X (gfx1250) — matrix operand order repaired and checked by reading, not compiled, not run on hardware
//
#include <hip/hip_runtime.h>
#include <hip/hip_bf16.h>
#include <math.h>

// ---- problem constants (match reference) ----
#define TOK 32      // B*S tokens
#define HDIM 2048
#define IDIM 5632
#define NEXP 8
#define SWIGLU_SCALE 1.702f

typedef __bf16 bf16_t;
typedef bf16_t v16bf __attribute__((ext_vector_type(16)));
typedef float  v8f   __attribute__((ext_vector_type(8)));

// ---------------------------------------------------------------------------
// WMMA helper: D = A(16x32 bf16) * B(32x16 bf16) + C(16x16 f32)
// ---------------------------------------------------------------------------
__device__ __forceinline__ v8f wmma_bf16(v16bf a, v16bf b, v8f c) {
    return __builtin_amdgcn_wmma_f32_16x16x32_bf16(
        /*neg_a=*/false, a, /*neg_b=*/false, b,
        /*c_mod=*/(short)0, c, /*reuse_a=*/false, /*reuse_b=*/false);
}

// A fragment: lane holds row M = lane%16 of a 16x32 tile.
// lanes 0-15: K = {k0..k0+15}; lanes 16-31: K = {k0+16..k0+31}
// rowPtr = &src[row * ld]; requires (k0 + half*16) float4-aligned (true: multiples of 16).
__device__ __forceinline__ v16bf load_a_frag(const float* __restrict__ rowPtr,
                                             int k0, int half) {
    const float4* p0 = reinterpret_cast<const float4*>(rowPtr + k0 + half * 16);
    const float4* p1 = reinterpret_cast<const float4*>(rowPtr + k0 + 8 + half * 16);
    float4 x0 = p0[0], x1 = p0[1];
    float4 y0 = p1[0], y1 = p1[1];
    v16bf a;
    a[0] = (bf16_t)x0.x; a[1] = (bf16_t)x0.y; a[2]  = (bf16_t)x0.z; a[3]  = (bf16_t)x0.w;
    a[4] = (bf16_t)x1.x; a[5] = (bf16_t)x1.y; a[6]  = (bf16_t)x1.z; a[7]  = (bf16_t)x1.w;
    a[8] = (bf16_t)y0.x; a[9] = (bf16_t)y0.y; a[10] = (bf16_t)y0.z; a[11] = (bf16_t)y0.w;
    a[12]= (bf16_t)y1.x; a[13]= (bf16_t)y1.y; a[14] = (bf16_t)y1.z; a[15] = (bf16_t)y1.w;
    return a;
}

// B fragment: lane holds column N = lane%16; element j = B[k0 + 16*(lane/16) + j][N].
// colPtr = &w[(k0 + 16*half) * ld + n0 + (lane%16)]. Strided per lane, coalesced
// across the 16 lanes of each half-wave (64B segments).
__device__ __forceinline__ v16bf load_b_frag(const float* __restrict__ colPtr,
                                             size_t ld) {
    v16bf b;
#pragma unroll
    for (int j = 0; j < 16; ++j) b[j] = (bf16_t)colPtr[(size_t)j * ld];
    return b;
}

// ---------------------------------------------------------------------------
// Kernel 1: router -> dense combine weights [TOK][NEXP] (zeros except top-2)
// ---------------------------------------------------------------------------
__global__ void moe_router_k(const float* __restrict__ x,
                             const float* __restrict__ rw,
                             float* __restrict__ comb) {
    __shared__ float logits[TOK][NEXP];
    int tid = threadIdx.x;             // 256 threads = 32 tokens x 8 experts
    int t = tid >> 3, e = tid & 7;
    const float4* xp = reinterpret_cast<const float4*>(x + (size_t)t * HDIM);
    const float4* wp = reinterpret_cast<const float4*>(rw + (size_t)e * HDIM);
    float acc = 0.f;
    for (int k = 0; k < HDIM / 4; ++k) {
        float4 a = xp[k], b = wp[k];
        acc += a.x * b.x + a.y * b.y + a.z * b.z + a.w * b.w;
    }
    logits[t][e] = acc;
    __syncthreads();
    if (tid < TOK) {
        float l[NEXP], a[NEXP];
        float mx = -INFINITY;
#pragma unroll
        for (int j = 0; j < NEXP; ++j) { l[j] = logits[tid][j]; mx = fmaxf(mx, l[j]); }
        float sum = 0.f;
#pragma unroll
        for (int j = 0; j < NEXP; ++j) { a[j] = __expf(l[j] - mx); sum += a[j]; }
        float inv = 1.f / sum;
#pragma unroll
        for (int j = 0; j < NEXP; ++j) a[j] *= inv;
        // top-2, ties broken toward lower index (matches jax.lax.top_k)
        int i0 = 0;
        for (int j = 1; j < NEXP; ++j) if (a[j] > a[i0]) i0 = j;
        int i1 = (i0 == 0) ? 1 : 0;
        for (int j = 0; j < NEXP; ++j) { if (j == i0) continue; if (a[j] > a[i1]) i1 = j; }
#pragma unroll
        for (int j = 0; j < NEXP; ++j)
            comb[tid * NEXP + j] = (j == i0 || j == i1) ? a[j] : 0.f;
    }
}

// ---------------------------------------------------------------------------
// Kernel 2: gate/up GEMMs + SwiGLU + combine-scale -> inter[E][TOK][IDIM] (f32)
// grid.x = NEXP * (IDIM/128); 256 threads = 8 waves, each wave: one 16-col
// N-tile, both 16-row M-tiles (weights read exactly once).
// ---------------------------------------------------------------------------
__global__ void moe_gateup_k(const float* __restrict__ x,
                             const float* __restrict__ wg,
                             const float* __restrict__ wu,
                             const float* __restrict__ comb,
                             float* __restrict__ inter) {
    const int IT = IDIM / 128;                       // 44 i-tiles
    int e  = blockIdx.x / IT;
    int i0 = (blockIdx.x % IT) * 128;
    int w    = threadIdx.x >> 5;                     // wave 0..7 (wave32)
    int lane = threadIdx.x & 31;
    int half = lane >> 4;
    int m    = lane & 15;
    int n    = lane & 15;
    int n0   = i0 + w * 16;

    const float* wgE  = wg + (size_t)e * HDIM * IDIM;
    const float* wuE  = wu + (size_t)e * HDIM * IDIM;
    const float* rowA0 = x + (size_t)m * HDIM;        // tokens 0..15
    const float* rowA1 = x + (size_t)(m + 16) * HDIM; // tokens 16..31
    const int koff = half * 16;

    v8f g0 = {0,0,0,0,0,0,0,0}, g1 = {0,0,0,0,0,0,0,0};
    v8f u0 = {0,0,0,0,0,0,0,0}, u1 = {0,0,0,0,0,0,0,0};

    for (int k0 = 0; k0 < HDIM; k0 += 32) {
        const float* pg = wgE + (size_t)(k0 + koff) * IDIM + n0 + n;
        const float* pu = wuE + (size_t)(k0 + koff) * IDIM + n0 + n;
        if (k0 + 32 < HDIM) {                        // stream-ahead hint
            __builtin_prefetch(pg + (size_t)32 * IDIM, 0, 1);
            __builtin_prefetch(pu + (size_t)32 * IDIM, 0, 1);
        }
        v16bf a0 = load_a_frag(rowA0, k0, half);
        v16bf a1 = load_a_frag(rowA1, k0, half);
        v16bf bg = load_b_frag(pg, IDIM);
        v16bf bu = load_b_frag(pu, IDIM);
        g0 = wmma_bf16(a0, bg, g0);
        g1 = wmma_bf16(a1, bg, g1);
        u0 = wmma_bf16(a0, bu, u0);
        u1 = wmma_bf16(a1, bu, u1);
    }

    int col = n0 + n;
#pragma unroll
    for (int r = 0; r < 8; ++r) {
        int t0 = half * 8 + r;                       // C layout: M = 8*(lane/16)+r
        int t1 = 16 + half * 8 + r;
        float c0 = comb[t0 * NEXP + e];
        float c1 = comb[t1 * NEXP + e];
        float gv = g0[r], uv = u0[r];
        float s  = gv / (1.f + __expf(-SWIGLU_SCALE * gv));
        inter[(size_t)(e * TOK + t0) * IDIM + col] = s * uv * c0;
        gv = g1[r]; uv = u1[r];
        s  = gv / (1.f + __expf(-SWIGLU_SCALE * gv));
        inter[(size_t)(e * TOK + t1) * IDIM + col] = s * uv * c1;
    }
}

// ---------------------------------------------------------------------------
// Kernel 3: down GEMM partials -> outp[16][TOK][HDIM]
// grid.x = NEXP * (HDIM/128) * 2 K-splits; each block contracts half of IDIM.
// ---------------------------------------------------------------------------
__global__ void moe_down_k(const float* __restrict__ inter,
                           const float* __restrict__ wd,
                           float* __restrict__ outp) {
    const int HT = HDIM / 128;                       // 16 h-tiles
    int e   = blockIdx.x / (HT * 2);
    int rem = blockIdx.x % (HT * 2);
    int h0  = (rem >> 1) * 128;
    int sp  = rem & 1;
    int kbeg = sp * (IDIM / 2), kend = kbeg + IDIM / 2;

    int w    = threadIdx.x >> 5;
    int lane = threadIdx.x & 31;
    int half = lane >> 4;
    int m    = lane & 15;
    int n    = lane & 15;
    const int koff = half * 16;

    const float* wdE   = wd + (size_t)e * IDIM * HDIM;
    const float* rowA0 = inter + (size_t)(e * TOK + m) * IDIM;
    const float* rowA1 = inter + (size_t)(e * TOK + m + 16) * IDIM;

    v8f c0 = {0,0,0,0,0,0,0,0}, c1 = {0,0,0,0,0,0,0,0};

    for (int k0 = kbeg; k0 < kend; k0 += 32) {
        const float* pb = wdE + (size_t)(k0 + koff) * HDIM + h0 + w * 16 + n;
        if (k0 + 32 < kend)
            __builtin_prefetch(pb + (size_t)32 * HDIM, 0, 1);
        v16bf a0 = load_a_frag(rowA0, k0, half);
        v16bf a1 = load_a_frag(rowA1, k0, half);
        v16bf b  = load_b_frag(pb, HDIM);
        c0 = wmma_bf16(a0, b, c0);
        c1 = wmma_bf16(a1, b, c1);
    }

    int slice = e * 2 + sp;                          // 0..15
    size_t base = (size_t)slice * TOK * HDIM;
    int col = h0 + w * 16 + n;
#pragma unroll
    for (int r = 0; r < 8; ++r) {
        int t0 = half * 8 + r;
        int t1 = 16 + half * 8 + r;
        outp[base + (size_t)t0 * HDIM + col] = c0[r];
        outp[base + (size_t)t1 * HDIM + col] = c1[r];
    }
}

// ---------------------------------------------------------------------------
// Kernel 4: deterministic fixed-order reduction of 16 partial slices
// ---------------------------------------------------------------------------
__global__ void moe_reduce_k(const float* __restrict__ outp,
                             float* __restrict__ out) {
    int i = blockIdx.x * blockDim.x + threadIdx.x;   // 0..TOK*HDIM-1
    float s = 0.f;
#pragma unroll
    for (int sl = 0; sl < 16; ++sl)
        s += outp[(size_t)sl * (TOK * HDIM) + i];
    out[i] = s;
}

// ---------------------------------------------------------------------------
extern "C" void kernel_launch(void* const* d_in, const int* in_sizes, int n_in,
                              void* d_out, int out_size, void* d_ws, size_t ws_size,
                              hipStream_t stream) {
    const float* x  = (const float*)d_in[0];   // [32,1,2048]
    const float* rw = (const float*)d_in[1];   // [8,2048]
    const float* wg = (const float*)d_in[2];   // [8,2048,5632]
    const float* wu = (const float*)d_in[3];   // [8,2048,5632]
    const float* wd = (const float*)d_in[4];   // [8,5632,2048]
    // d_in[5] = top_k (always 2 in this configuration)
    float* out = (float*)d_out;

    // workspace layout (offsets keep 16B alignment for float4 loads)
    float* comb  = (float*)d_ws;                         // 256 f32 (padded to 1KB)
    float* inter = comb + 256;                           // NEXP*TOK*IDIM f32
    float* outp  = inter + (size_t)NEXP * TOK * IDIM;    // 16*TOK*HDIM f32

    moe_router_k<<<1, 256, 0, stream>>>(x, rw, comb);
    moe_gateup_k<<<NEXP * (IDIM / 128), 256, 0, stream>>>(x, wg, wu, comb, inter);
    moe_down_k<<<NEXP * (HDIM / 128) * 2, 256, 0, stream>>>(inter, wd, outp);
    moe_reduce_k<<<(TOK * HDIM) / 256, 256, 0, stream>>>(outp, out);
}
